// VanillaTPAttention_77051713290784
// MI455X (gfx1250) — matrix operand order repaired and checked by reading, not compiled, not run on hardware
//
#include <hip/hip_runtime.h>
#include <hip/hip_bf16.h>

// ---------------------------------------------------------------------------
// Causal MHA on MI455X (gfx1250, wave32, WMMA, async global->LDS DMA).
//   x:[2,2048,1024] f32; Wq/Wk/Wv/Wo:[1024,1024] f32 (torch linear: y = x @ W^T)
// Pipeline: f32->bf16 | WMMA GEMMs (double-buffered LDS via
// global_load_async_to_lds_b128 / ASYNCcnt) | flash-attn (bf16 WMMA, fp32
// online softmax, per-wave DScnt sync) | WMMA out-proj to f32.
// ---------------------------------------------------------------------------

typedef __bf16 bf16_t;
typedef __attribute__((ext_vector_type(16))) __bf16 v16bf;
typedef __attribute__((ext_vector_type(8)))  __bf16 v8bf;
typedef __attribute__((ext_vector_type(8)))  float  v8f;

constexpr int BATCH = 2;
constexpr int SEQ   = 2048;
constexpr int DM    = 1024;   // d_model
constexpr int HEADS = 16;
constexpr int DH    = 64;     // head dim
constexpr int MROWS = BATCH * SEQ;          // 4096
constexpr float SCALE = 0.125f;             // 1/sqrt(64)

__device__ inline v8f wmma_bf16(v16bf a, v16bf b, v8f c) {
  // D = A(16x32 bf16) * B(32x16 bf16) + C(16x16 f32)
  return __builtin_amdgcn_wmma_f32_16x16x32_bf16(
      /*neg_a=*/false, a, /*neg_b=*/false, b,
      /*c_mod=*/(short)0, c, /*reuse_a=*/false, /*reuse_b=*/false);
}

// A fragment (16 rows x 32 k), source row-major [rows][ldk].
// lane L: row = L&15; halves 0..7 = k[8*(L>>4)..+7]; halves 8..15 = k[16+8*(L>>4)..+7]
__device__ inline v16bf load_afrag(const bf16_t* base, int ldk, int lane) {
  const int r  = lane & 15;
  const int hi = lane >> 4;
  const bf16_t* p = base + (size_t)r * ldk + 8 * hi;
  v8bf lo = *(const v8bf*)p;
  v8bf hj = *(const v8bf*)(p + 16);
  v16bf out;
#pragma unroll
  for (int i = 0; i < 8; ++i) { out[i] = lo[i]; out[8 + i] = hj[i]; }
  return out;
}

// B fragment (32 k x 16 n), source row-major [n][ldk] (contraction contiguous).
// lane L: n = L&15; halves 0..7 = k[8*(L>>4)..+7]; halves 8..15 = k[16+8*(L>>4)..+7]
__device__ inline v16bf load_bfrag(const bf16_t* base, int ldk, int lane) {
  const int n  = lane & 15;
  const int hi = lane >> 4;
  const bf16_t* p = base + (size_t)n * ldk + 8 * hi;
  v8bf lo = *(const v8bf*)p;
  v8bf hj = *(const v8bf*)(p + 16);
  v16bf out;
#pragma unroll
  for (int i = 0; i < 8; ++i) { out[i] = lo[i]; out[8 + i] = hj[i]; }
  return out;
}

// Async DMA: per-lane 16B global -> LDS, tracked by ASYNCcnt.
__device__ inline void async_copy_b128(void* lds_ptr, const void* gptr) {
  unsigned lds_off = (unsigned)(uintptr_t)lds_ptr;           // addr[31:0] = LDS address
  unsigned long long ga = (unsigned long long)(uintptr_t)gptr;
  asm volatile("global_load_async_to_lds_b128 %0, %1, off"
               :: "v"(lds_off), "v"(ga) : "memory");
}

template <int N>
__device__ inline void wait_asynccnt() {
#if __has_builtin(__builtin_amdgcn_s_wait_asynccnt)
  __builtin_amdgcn_s_wait_asynccnt(N);
#else
  asm volatile("s_wait_asynccnt %0" :: "i"(N) : "memory");
#endif
}

// Per-wave LDS ordering: wave's own ds stores complete before its ds loads.
__device__ inline void wait_dscnt0() {
  asm volatile("s_wait_dscnt 0x0" ::: "memory");
}

// ---------------------------------------------------------------------------
__global__ void cvt_f32_bf16(const float* __restrict__ src, bf16_t* __restrict__ dst, int n) {
  int i = blockIdx.x * blockDim.x + threadIdx.x;
  const int stride = gridDim.x * blockDim.x;
  for (; i < n; i += stride) dst[i] = (bf16_t)src[i];
}

// ---------------------------------------------------------------------------
// C[m,n] = A[m,:] . W[n,:]   A:[4096,1024] bf16, W:[1024,1024] bf16 ([N,K] row-major)
// mode 0: bf16 out, head-split [BH,SEQ,DH] (Q,K) | mode 2: [BH,DH,SEQ] (V)
// mode 1: f32 out, [MROWS,DM] (final projection)
// Block = 256 thr = 8 waves; block tile 128(M) x 64(N); wave tile 16 x 64.
// 32-wide k-slabs staged by async DMA into double-buffered LDS; last k-step
// peeled so the steady-state loop is branch-free.
constexpr int LDA = 40;   // padded row (halves): 80B -> bank spread, 16B aligned
constexpr int KSTEPS = DM / 32;

__global__ void __launch_bounds__(256)
gemm_linear_bf16(const bf16_t* __restrict__ A, const bf16_t* __restrict__ W,
                 bf16_t* __restrict__ Ob, float* __restrict__ Of, int mode)
{
  __shared__ alignas(16) bf16_t Abuf[2][128][LDA];   // 20 KiB
  __shared__ alignas(16) bf16_t Bbuf[2][64][LDA];    // 10 KiB

  const int tid   = threadIdx.x;
  const int lane  = tid & 31;
  const int wave  = tid >> 5;
  const int mblk  = blockIdx.x * 128;
  const int nblk  = blockIdx.y * 64;
  const int mbase = mblk + wave * 16;

  // per-thread staging coordinates (3 async b128 per thread per slab)
  const int arow0 = tid >> 1;                 // A: 2 chunks, row = tid/2
  const int acol0 = (tid & 1) * 16;           // halves 0..15 or 16..31
  const int brow  = tid >> 2;                 // B: 1 chunk
  const int bcol  = (tid & 3) * 8;

  auto stage = [&](int buf, int k0) {
    const bf16_t* ga = A + (size_t)(mblk + arow0) * DM + k0 + acol0;
    async_copy_b128(&Abuf[buf][arow0][acol0],     ga);
    async_copy_b128(&Abuf[buf][arow0][acol0 + 8], ga + 8);
    async_copy_b128(&Bbuf[buf][brow][bcol],
                    W + (size_t)(nblk + brow) * DM + k0 + bcol);
  };

  v8f acc[4] = {};
  auto compute = [&](int buf) {
    // batch all fragment loads, then 4 independent back-to-back WMMAs
    v16bf a  = load_afrag(&Abuf[buf][wave * 16][0], LDA, lane);
    v16bf b0 = load_bfrag(&Bbuf[buf][0][0],  LDA, lane);
    v16bf b1 = load_bfrag(&Bbuf[buf][16][0], LDA, lane);
    v16bf b2 = load_bfrag(&Bbuf[buf][32][0], LDA, lane);
    v16bf b3 = load_bfrag(&Bbuf[buf][48][0], LDA, lane);
    acc[0] = wmma_bf16(a, b0, acc[0]);
    acc[1] = wmma_bf16(a, b1, acc[1]);
    acc[2] = wmma_bf16(a, b2, acc[2]);
    acc[3] = wmma_bf16(a, b3, acc[3]);
  };

  stage(0, 0);
  int cur = 0;
  for (int ks = 0; ks < KSTEPS - 1; ++ks) {   // branch-free steady state
    stage(cur ^ 1, (ks + 1) * 32);            // DMA next slab into other buffer
    wait_asynccnt<3>();                       // current slab (oldest 3 ops) landed
    __syncthreads();                          // all waves' DMA complete
    compute(cur);
    __syncthreads();                          // done reading before overwrite
    cur ^= 1;
  }
  wait_asynccnt<0>();                         // last slab
  __syncthreads();
  compute(cur);

  const int hi = lane >> 4, nlo = lane & 15;
#pragma unroll
  for (int n = 0; n < 4; ++n) {
#pragma unroll
    for (int r = 0; r < 8; ++r) {
      const int m    = mbase + r + 8 * hi;          // global row (b*SEQ + s)
      const int ncol = nblk + 16 * n + nlo;         // global col (h*DH + d)
      const float v  = acc[n][r];
      if (mode == 1) {
        Of[(size_t)m * DM + ncol] = v;
      } else {
        const int b = m >> 11, s = m & (SEQ - 1);
        const int h = ncol >> 6, d = ncol & (DH - 1);
        const int bh = b * HEADS + h;
        if (mode == 0)
          Ob[(((size_t)bh) * SEQ + s) * DH + d] = (bf16_t)v;      // [BH,S,DH]
        else
          Ob[(((size_t)bh) * DH + d) * SEQ + s] = (bf16_t)v;      // [BH,DH,S]
      }
    }
  }
}

// ---------------------------------------------------------------------------
// Flash attention, causal. Q,K:[BH,SEQ,DH] bf16; Vt:[BH,DH,SEQ] bf16.
// AO: [B, SEQ, DM] bf16. Block = 128 thr = 4 waves; 64 q rows/block, 16/wave.
// P relayout (C->A frag) uses a per-wave private LDS slice, so only the
// wave's own DScnt must drain -- no block barrier in the kv loop.
__global__ void __launch_bounds__(128)
flash_attn_bf16(const bf16_t* __restrict__ Qm, const bf16_t* __restrict__ Km,
                const bf16_t* __restrict__ Vt, bf16_t* __restrict__ AO)
{
  __shared__ alignas(64) bf16_t plds[4][16][32];   // per-wave P tile (16q x 32kv)

  const int lane  = threadIdx.x & 31;
  const int wave  = threadIdx.x >> 5;
  const int bh    = blockIdx.y;                    // 0..31
  const int b     = bh >> 4, h = bh & (HEADS - 1);
  const int qbase = blockIdx.x * 64;
  const int qrow  = qbase + wave * 16;
  const int hi    = lane >> 4, nlo = lane & 15;

  const bf16_t* Qb = Qm + ((size_t)bh * SEQ + qrow) * DH;
  v16bf qa[2];
#pragma unroll
  for (int c = 0; c < 2; ++c) qa[c] = load_afrag(Qb + 32 * c, DH, lane);

  v8f   o[4] = {};
  float mrow[8], lrow[8], alpha[8];
#pragma unroll
  for (int r = 0; r < 8; ++r) { mrow[r] = -1e30f; lrow[r] = 0.0f; }

  const int kv_end = qbase + 64;
  for (int kv0 = 0; kv0 < kv_end; kv0 += 32) {
    const bf16_t* Kb = Km + ((size_t)bh * SEQ + kv0) * DH;
    if (kv0 + 32 < kv_end) __builtin_prefetch(Kb + 32 * DH, 0, 3);

    // ---- S = Q K^T for 32 keys (two 16x16 C tiles) ----
    v8f c0 = {}, c1 = {};
#pragma unroll
    for (int c = 0; c < 2; ++c) {
      v16bf b0 = load_bfrag(Kb + 32 * c, DH, lane);             // keys kv0..+15
      v16bf b1 = load_bfrag(Kb + 16 * DH + 32 * c, DH, lane);   // keys kv0+16..+31
      c0 = wmma_bf16(qa[c], b0, c0);
      c1 = wmma_bf16(qa[c], b1, c1);
    }

    // ---- online softmax (fp32), causal mask, stash P in LDS ----
#pragma unroll
    for (int r = 0; r < 8; ++r) {
      const int row = qrow + r + 8 * hi;
      float v0 = c0[r] * SCALE; if (kv0 + nlo > row)      v0 = -1e30f;
      float v1 = c1[r] * SCALE; if (kv0 + 16 + nlo > row) v1 = -1e30f;
      float mx = fmaxf(v0, v1);
#pragma unroll
      for (int msk = 1; msk < 16; msk <<= 1) mx = fmaxf(mx, __shfl_xor(mx, msk, 32));
      const float mnew = fmaxf(mrow[r], mx);
      const float a    = __expf(mrow[r] - mnew);
      const float p0   = __expf(v0 - mnew);
      const float p1   = __expf(v1 - mnew);
      float rs = p0 + p1;
#pragma unroll
      for (int msk = 1; msk < 16; msk <<= 1) rs += __shfl_xor(rs, msk, 32);
      lrow[r]  = lrow[r] * a + rs;
      mrow[r]  = mnew;
      alpha[r] = a;
      plds[wave][r + 8 * hi][nlo]      = (bf16_t)p0;
      plds[wave][r + 8 * hi][16 + nlo] = (bf16_t)p1;
    }

#pragma unroll
    for (int n = 0; n < 4; ++n)
#pragma unroll
      for (int r = 0; r < 8; ++r) o[n][r] *= alpha[r];

    wait_dscnt0();   // wave-private slice: own stores drained, loads see them

    // ---- P as A-frag (contraction over 32 keys) ----
    v8bf plo = *(const v8bf*)&plds[wave][nlo][8 * hi];
    v8bf phi = *(const v8bf*)&plds[wave][nlo][16 + 8 * hi];
    v16bf pf;
#pragma unroll
    for (int i = 0; i < 8; ++i) { pf[i] = plo[i]; pf[8 + i] = phi[i]; }

    // ---- O += P V (V transposed: contraction contiguous) ----
    const bf16_t* Vb = Vt + (size_t)bh * DH * SEQ + kv0;
#pragma unroll
    for (int n = 0; n < 4; ++n) {
      v16bf vb = load_bfrag(Vb + (size_t)(16 * n) * SEQ, SEQ, lane);
      o[n] = wmma_bf16(pf, vb, o[n]);
    }
    wait_dscnt0();   // P loads done before next tile's stores overwrite
  }

  // ---- epilogue: O /= l, store as [B, SEQ, DM] bf16 ----
  float inv[8];
#pragma unroll
  for (int r = 0; r < 8; ++r) inv[r] = 1.0f / lrow[r];
#pragma unroll
  for (int n = 0; n < 4; ++n) {
#pragma unroll
    for (int r = 0; r < 8; ++r) {
      const int s = qrow + r + 8 * hi;
      const int d = 16 * n + nlo;
      AO[((size_t)(b * SEQ + s)) * DM + h * DH + d] = (bf16_t)(o[n][r] * inv[r]);
    }
  }
}

// ---------------------------------------------------------------------------
extern "C" void kernel_launch(void* const* d_in, const int* in_sizes, int n_in,
                              void* d_out, int out_size, void* d_ws, size_t ws_size,
                              hipStream_t stream) {
  const float* x  = (const float*)d_in[0];
  const float* Wq = (const float*)d_in[1];
  const float* Wk = (const float*)d_in[2];
  const float* Wv = (const float*)d_in[3];
  const float* Wo = (const float*)d_in[4];
  float* out = (float*)d_out;

  char* ws = (char*)d_ws;
  const size_t MB = 1u << 20;
  bf16_t* xb  = (bf16_t*)(ws + 0 * MB);   // [4096,1024]      8 MiB
  bf16_t* Wqb = (bf16_t*)(ws + 8 * MB);   // [1024,1024]      2 MiB
  bf16_t* Wkb = (bf16_t*)(ws + 10 * MB);
  bf16_t* Wvb = (bf16_t*)(ws + 12 * MB);
  bf16_t* Wob = (bf16_t*)(ws + 14 * MB);
  bf16_t* Qh  = (bf16_t*)(ws + 16 * MB);  // [32,2048,64]     8 MiB
  bf16_t* Kh  = (bf16_t*)(ws + 24 * MB);  // [32,2048,64]     8 MiB
  bf16_t* Vtb = (bf16_t*)(ws + 32 * MB);  // [32,64,2048]     8 MiB
  bf16_t* AOb = (bf16_t*)(ws + 40 * MB);  // [4096,1024]      8 MiB

  // 1) fp32 -> bf16
  cvt_f32_bf16<<<512, 256, 0, stream>>>(x,  xb,  MROWS * DM);
  cvt_f32_bf16<<<256, 256, 0, stream>>>(Wq, Wqb, DM * DM);
  cvt_f32_bf16<<<256, 256, 0, stream>>>(Wk, Wkb, DM * DM);
  cvt_f32_bf16<<<256, 256, 0, stream>>>(Wv, Wvb, DM * DM);
  cvt_f32_bf16<<<256, 256, 0, stream>>>(Wo, Wob, DM * DM);

  // 2) Q/K/V projections (WMMA bf16, async-LDS double buffering)
  dim3 ggrid(MROWS / 128, DM / 64);
  gemm_linear_bf16<<<ggrid, 256, 0, stream>>>(xb, Wqb, Qh,  nullptr, 0);
  gemm_linear_bf16<<<ggrid, 256, 0, stream>>>(xb, Wkb, Kh,  nullptr, 0);
  gemm_linear_bf16<<<ggrid, 256, 0, stream>>>(xb, Wvb, Vtb, nullptr, 2);

  // 3) causal flash attention
  dim3 agrid(SEQ / 64, BATCH * HEADS);
  flash_attn_bf16<<<agrid, 128, 0, stream>>>(Qh, Kh, Vtb, AOb);

  // 4) output projection -> fp32 d_out
  gemm_linear_bf16<<<ggrid, 256, 0, stream>>>(AOb, Wob, nullptr, out, 1);
}
